// Attention_70677981823488
// MI455X (gfx1250) — compile-verified, matrix operand order checked by reading, not run on hardware
//
#include <hip/hip_runtime.h>

#define HEADS  16
#define DHEAD  64
#define NTOK   2048
#define BATCH  2
#define DIM    1024
#define INNER  1024
#define ROT    32

typedef __attribute__((ext_vector_type(16))) _Float16 v16h;
typedef __attribute__((ext_vector_type(8)))  float    v8f;

template <bool B> struct bool_c { static constexpr bool value = B; };

static __device__ inline v16h lds16(const _Float16* p) {
  return *(const v16h*)p;
}

static __device__ inline v8f wmma_f16(v16h a, v16h b, v8f c) {
  return __builtin_amdgcn_wmma_f32_16x16x32_f16(false, a, false, b, (short)0, c,
                                                false, false);
}

// Low 32 bits of a generic pointer into LDS == wave-relative LDS byte address
// (flat->LDS aperture mapping uses addr[31:0], CDNA5 ISA 10.2).
static __device__ inline unsigned lds_addr(const void* p) {
  return (unsigned)(unsigned long long)p;
}

// CDNA5 async global->LDS copy (16B per lane), tracked by ASYNCcnt.
static __device__ inline void async_g2l_b128(unsigned ldsaddr, const void* gaddr) {
  asm volatile("global_load_async_to_lds_b128 %0, %1, off"
               :: "v"(ldsaddr), "v"((unsigned long long)gaddr)
               : "memory");
}

static __device__ inline void wait_async0() {
  asm volatile("s_wait_asynccnt 0" ::: "memory");
}

static __device__ inline void wait_ds0() {
  asm volatile("s_wait_dscnt 0" ::: "memory");
}

// xor-butterfly max within 16-lane groups, on the VALU (v_permlane16_b32)
// instead of the LDS pipe (ds_bpermute), falling back to __shfl_xor.
static __device__ inline float grpmax16(float v) {
#if __has_builtin(__builtin_amdgcn_permlane16)
  int x = __float_as_int(v);
  float t;
  t = __int_as_float(__builtin_amdgcn_permlane16(x, x, 0x67452301, 0xEFCDAB89, false, false)); // ^1
  v = fmaxf(v, t); x = __float_as_int(v);
  t = __int_as_float(__builtin_amdgcn_permlane16(x, x, 0x54761032, 0xDCFE98BA, false, false)); // ^2
  v = fmaxf(v, t); x = __float_as_int(v);
  t = __int_as_float(__builtin_amdgcn_permlane16(x, x, 0x32107654, 0xBA98FEDC, false, false)); // ^4
  v = fmaxf(v, t); x = __float_as_int(v);
  t = __int_as_float(__builtin_amdgcn_permlane16(x, x, 0xFEDCBA98, 0x76543210, false, false)); // ^8
  v = fmaxf(v, t);
  return v;
#else
#pragma unroll
  for (int off = 8; off >= 1; off >>= 1)
    v = fmaxf(v, __shfl_xor(v, off, 32));
  return v;
#endif
}

// ---------------- L2 norm (reference: x / max(||x||/sqrt(dim), eps) * gamma) --
__global__ __launch_bounds__(256) void rmsnorm_f16(const float* __restrict__ x,
                                                   const float* __restrict__ gamma,
                                                   _Float16* __restrict__ xn) {
  __shared__ float red[256];
  int row = blockIdx.x;
  const float* xr = x + (size_t)row * DIM;
  float ss = 0.f;
  for (int i = threadIdx.x; i < DIM; i += 256) { float v = xr[i]; ss += v * v; }
  red[threadIdx.x] = ss;
  __syncthreads();
  for (int s = 128; s > 0; s >>= 1) {
    if (threadIdx.x < s) red[threadIdx.x] += red[threadIdx.x + s];
    __syncthreads();
  }
  float norm = sqrtf(red[0] * (1.0f / DIM));
  float rinv = 1.0f / fmaxf(norm, 1e-8f);
  for (int i = threadIdx.x; i < DIM; i += 256)
    xn[(size_t)row * DIM + i] = (_Float16)(xr[i] * rinv * gamma[i]);
}

// ------------- weight convert+transpose: W(K x N) f32 -> Wt(N x K) f16 -------
__global__ __launch_bounds__(256) void transpose_to_f16(const float* __restrict__ W,
                                                        _Float16* __restrict__ Wt,
                                                        int K, int N) {
  int k = blockIdx.x * 32 + threadIdx.x;
  int n = blockIdx.y * 8 + threadIdx.y;
  if (k < K && n < N)
    Wt[(size_t)n * K + k] = (_Float16)W[(size_t)k * N + n];
}

// -------- LDS double-buffered WMMA GEMM: C(MxN) = A(MxK) @ Bt(NxK)^T ---------
// block: 256 threads = 8 waves, tile 128(M) x 64(N), k-step 32, async staging.
template <bool F16OUT>
__global__ __launch_bounds__(256) void gemm_lds(const _Float16* __restrict__ A,
                                                const _Float16* __restrict__ Bt,
                                                _Float16* __restrict__ Ch,
                                                float* __restrict__ Cf,
                                                const float* __restrict__ bias,
                                                int M, int N, int K) {
  __shared__ _Float16 As[2][128][32];  // 2 x 8KB
  __shared__ _Float16 Bs[2][64][32];   // 2 x 4KB
  int tid = threadIdx.x;
  int lane = tid & 31;
  int wave = tid >> 5;
  int tm0 = blockIdx.x * 128;
  int tn0 = blockIdx.y * 64;
  int khalf = (lane >> 4) * 16;

  // per-thread async chunk assignment (16B chunks)
  int rA0 = tid >> 2,          sA0 = tid & 3;           // A chunks 0..255
  int rA1 = (tid + 256) >> 2,  sA1 = tid & 3;           // A chunks 256..511
  int rB  = tid >> 2,          sB  = tid & 3;           // B chunks 0..255

  v8f acc[4] = {};

  const int NK = K / 32;
  // prologue: fill buffer 0
  {
    async_g2l_b128(lds_addr(&As[0][rA0][sA0 * 8]),
                   A + (size_t)(tm0 + rA0) * K + sA0 * 8);
    async_g2l_b128(lds_addr(&As[0][rA1][sA1 * 8]),
                   A + (size_t)(tm0 + rA1) * K + sA1 * 8);
    async_g2l_b128(lds_addr(&Bs[0][rB][sB * 8]),
                   Bt + (size_t)(tn0 + rB) * K + sB * 8);
  }
  wait_async0();
  __syncthreads();

  for (int ks = 0; ks < NK; ++ks) {
    int cur = ks & 1;
    if (ks + 1 < NK) {
      int nk = (ks + 1) * 32;
      int nb = cur ^ 1;
      wait_ds0();
      async_g2l_b128(lds_addr(&As[nb][rA0][sA0 * 8]),
                     A + (size_t)(tm0 + rA0) * K + nk + sA0 * 8);
      async_g2l_b128(lds_addr(&As[nb][rA1][sA1 * 8]),
                     A + (size_t)(tm0 + rA1) * K + nk + sA1 * 8);
      async_g2l_b128(lds_addr(&Bs[nb][rB][sB * 8]),
                     Bt + (size_t)(tn0 + rB) * K + nk + sB * 8);
    }
    // compute from LDS buffer `cur`
    v16h a  = lds16(&As[cur][wave * 16 + (lane & 15)][khalf]);
    v16h b0 = lds16(&Bs[cur][0  + (lane & 15)][khalf]);
    v16h b1 = lds16(&Bs[cur][16 + (lane & 15)][khalf]);
    v16h b2 = lds16(&Bs[cur][32 + (lane & 15)][khalf]);
    v16h b3 = lds16(&Bs[cur][48 + (lane & 15)][khalf]);
    acc[0] = wmma_f16(a, b0, acc[0]);
    acc[1] = wmma_f16(a, b1, acc[1]);
    acc[2] = wmma_f16(a, b2, acc[2]);
    acc[3] = wmma_f16(a, b3, acc[3]);
    if (ks + 1 < NK) wait_async0();
    __syncthreads();
  }

  int tm = tm0 + wave * 16;
  int rbase = (lane < 16) ? 0 : 8;
  int cn = lane & 15;
#pragma unroll
  for (int t = 0; t < 4; ++t)
#pragma unroll
    for (int r = 0; r < 8; ++r) {
      int col = tn0 + t * 16 + cn;
      size_t idx = (size_t)(tm + rbase + r) * N + col;
      if (F16OUT) Ch[idx] = (_Float16)acc[t][r];
      else        Cf[idx] = acc[t][r] + bias[col];
    }
}

// ---- RoPE + scale, (b,n,h*d) f16 -> (b,h,n,d) f16 ---------------------------
__global__ __launch_bounds__(256) void rope_head(const _Float16* __restrict__ lin,
                                                 const float* __restrict__ freqs,
                                                 _Float16* __restrict__ dst,
                                                 int colOff, int linStride, float scl) {
  int idx = blockIdx.x * 256 + threadIdx.x;  // b:1 h:4 n:11 d:6 bits
  int d  = idx & 63;
  int nn = (idx >> 6) & (NTOK - 1);
  int h  = (idx >> 17) & (HEADS - 1);
  int bb = idx >> 21;
  size_t src = (size_t)(bb * NTOK + nn) * linStride + colOff + h * DHEAD;
  float v = (float)lin[src + d];
  float out;
  if (d < ROT) {
    float f = freqs[nn * ROT + d];
    int dp = (d < 16) ? d + 16 : d - 16;
    float v2 = (float)lin[src + dp];
    float rh = (d < 16) ? -v2 : v2;
    out = v * cosf(f) + rh * sinf(f);
  } else {
    out = v;
  }
  dst[(size_t)idx] = (_Float16)(out * scl);
}

// ---- V: (b,n,2*inner) cols [inner,2*inner) -> Vt (b,h,d,n) f16 --------------
__global__ __launch_bounds__(256) void v_transpose(const _Float16* __restrict__ lin,
                                                   _Float16* __restrict__ vt) {
  int idx = blockIdx.x * 256 + threadIdx.x;  // b:1 h:4 d:6 n:11 bits
  int nn = idx & (NTOK - 1);
  int d  = (idx >> 11) & 63;
  int h  = (idx >> 17) & (HEADS - 1);
  int bb = idx >> 21;
  vt[(size_t)idx] =
      lin[(size_t)(bb * NTOK + nn) * (2 * INNER) + INNER + h * DHEAD + d];
}

// ---- flash attention, causal; async double-buffered K/V tiles in LDS --------
// - row-sum of P via extra WMMA against all-ones B (lands in C-frag layout)
// - row-max via VALU permlane16 butterfly
// - causal mask peeled: only the last key tile of each query block needs it
__global__ __launch_bounds__(32) void attn_fa(const _Float16* __restrict__ Q,
                                              const _Float16* __restrict__ Kh,
                                              const _Float16* __restrict__ Vt,
                                              _Float16* __restrict__ O) {
  __shared__ _Float16 Ks[2][32][64];  // 2 x 4KB, keys x d
  __shared__ _Float16 Vs[2][64][32];  // 2 x 4KB, d x keys
  __shared__ _Float16 P[16][32];
  int lane = threadIdx.x;
  int qi = blockIdx.x;
  int h  = blockIdx.y;
  int bb = blockIdx.z;
  int bh = bb * HEADS + h;
  const _Float16* q  = Q  + (size_t)bh * NTOK * DHEAD;
  const _Float16* kk = Kh + (size_t)bh * NTOK * DHEAD;
  const _Float16* vt = Vt + (size_t)bh * DHEAD * NTOK;
  int q0 = qi * 16;
  int rowm = lane & 15;
  int khalf = (lane >> 4) * 16;
  int rbase = (lane < 16) ? 0 : 8;

  v16h qa0 = lds16(q + (size_t)(q0 + rowm) * DHEAD + khalf);
  v16h qa1 = lds16(q + (size_t)(q0 + rowm) * DHEAD + 32 + khalf);

  v16h ones;
#pragma unroll
  for (int i = 0; i < 16; ++i) ones[i] = (_Float16)1.0f;

  float m[8];
  v8f o[4] = {};
  v8f lacc = {};
#pragma unroll
  for (int r = 0; r < 8; ++r) m[r] = -1e30f;

  int qend = q0 + 15;
  int niter = qend / 32 + 1;

  // prologue: stage K/V tile for kj = 0 into buffer 0 (16B chunks, 8+8/lane)
#pragma unroll
  for (int i = 0; i < 8; ++i) {
    int c = lane + 32 * i;
    int rk = c >> 3, sk = c & 7;
    async_g2l_b128(lds_addr(&Ks[0][rk][sk * 8]),
                   kk + (size_t)rk * DHEAD + sk * 8);
    int rv = c >> 2, sv = c & 3;
    async_g2l_b128(lds_addr(&Vs[0][rv][sv * 8]),
                   vt + (size_t)rv * NTOK + sv * 8);
  }
  wait_async0();

  auto step = [&](int it, auto maskedC) {
    constexpr bool MASKED = decltype(maskedC)::value;
    int kj = it * 32;
    int cur = it & 1;
    if (it + 1 < niter) {
      int nj = kj + 32;
      int nb = cur ^ 1;
      wait_ds0();
#pragma unroll
      for (int i = 0; i < 8; ++i) {
        int c = lane + 32 * i;
        int rk = c >> 3, sk = c & 7;
        async_g2l_b128(lds_addr(&Ks[nb][rk][sk * 8]),
                       kk + (size_t)(nj + rk) * DHEAD + sk * 8);
        int rv = c >> 2, sv = c & 3;
        async_g2l_b128(lds_addr(&Vs[nb][rv][sv * 8]),
                       vt + (size_t)rv * NTOK + nj + sv * 8);
      }
    }

    // S = Q K^T for keys kj..kj+31 (two 16x16 tiles)
    v8f s0 = {}, s1 = {};
    {
      v16h b0 = lds16(&Ks[cur][rowm][khalf]);
      v16h b1 = lds16(&Ks[cur][rowm][32 + khalf]);
      s0 = wmma_f16(qa0, b0, s0);
      s0 = wmma_f16(qa1, b1, s0);
      v16h c0 = lds16(&Ks[cur][16 + rowm][khalf]);
      v16h c1 = lds16(&Ks[cur][16 + rowm][32 + khalf]);
      s1 = wmma_f16(qa0, c0, s1);
      s1 = wmma_f16(qa1, c1, s1);
    }

    // (masked only in the last tile) + online softmax
    int col0 = kj + (lane & 15);
    int col1 = col0 + 16;
    float p0v[8], p1v[8], alpha[8];
#pragma unroll
    for (int r = 0; r < 8; ++r) {
      float a0 = s0[r];
      float a1 = s1[r];
      if (MASKED) {
        int qr = q0 + rbase + r;
        a0 = (col0 <= qr) ? a0 : -1e30f;
        a1 = (col1 <= qr) ? a1 : -1e30f;
      }
      float vm = grpmax16(fmaxf(a0, a1));
      float mn = fmaxf(m[r], vm);
      alpha[r] = __expf(m[r] - mn);
      p0v[r] = __expf(a0 - mn);
      p1v[r] = __expf(a1 - mn);
      m[r] = mn;
    }
#pragma unroll
    for (int r = 0; r < 8; ++r) {
      lacc[r] *= alpha[r];
#pragma unroll
      for (int t = 0; t < 4; ++t) o[t][r] *= alpha[r];
    }

    // restage P (C-fragment layout) into A-fragment layout through LDS
    __syncthreads();
    {
      int cw = lane & 15;
#pragma unroll
      for (int r = 0; r < 8; ++r) {
        P[rbase + r][cw]      = (_Float16)p0v[r];
        P[rbase + r][cw + 16] = (_Float16)p1v[r];
      }
    }
    __syncthreads();
    v16h pa = lds16(&P[rowm][khalf]);
    lacc = wmma_f16(pa, ones, lacc);  // row-sum(P) accumulated into l
#pragma unroll
    for (int t = 0; t < 4; ++t) {
      v16h vb = lds16(&Vs[cur][t * 16 + rowm][khalf]);
      o[t] = wmma_f16(pa, vb, o[t]);
    }
    if (it + 1 < niter) wait_async0();
  };

  // all-but-last key tiles are strictly below the diagonal: no mask VALU
  for (int it = 0; it + 1 < niter; ++it) step(it, bool_c<false>{});
  step(niter - 1, bool_c<true>{});

#pragma unroll
  for (int t = 0; t < 4; ++t)
#pragma unroll
    for (int r = 0; r < 8; ++r) {
      int row = q0 + rbase + r;
      int dcol = t * 16 + (lane & 15);
      O[(size_t)(bb * NTOK + row) * INNER + h * DHEAD + dcol] =
          (_Float16)(o[t][r] / lacc[r]);
    }
}

extern "C" void kernel_launch(void* const* d_in, const int* in_sizes, int n_in,
                              void* d_out, int out_size, void* d_ws, size_t ws_size,
                              hipStream_t stream) {
  const float* x     = (const float*)d_in[0];
  const float* freqs = (const float*)d_in[1];
  const float* gamma = (const float*)d_in[2];
  const float* Wq    = (const float*)d_in[3];
  const float* Wkv   = (const float*)d_in[4];
  const float* Wo    = (const float*)d_in[5];
  const float* bo    = (const float*)d_in[6];
  float* out = (float*)d_out;

  const size_t M = (size_t)BATCH * NTOK;  // 4096
  size_t need = 0;
  need += M * DIM;                 // xn
  need += (size_t)DIM * INNER;     // Wq_t
  need += (size_t)DIM * 2 * INNER; // Wkv_t
  need += (size_t)INNER * DIM;     // Wo_t
  need += M * INNER;               // q_lin (aliased later as attn out)
  need += M * 2 * INNER;           // kv_lin
  need += M * INNER * 3;           // q_arr, k_arr, v_t
  if (ws_size < need * sizeof(_Float16)) return;

  _Float16* p = (_Float16*)d_ws;
  _Float16* xn    = p; p += M * DIM;
  _Float16* wqt   = p; p += (size_t)DIM * INNER;
  _Float16* wkvt  = p; p += (size_t)DIM * 2 * INNER;
  _Float16* wot   = p; p += (size_t)INNER * DIM;
  _Float16* qlin  = p; p += M * INNER;
  _Float16* kvlin = p; p += M * 2 * INNER;
  _Float16* qarr  = p; p += M * INNER;
  _Float16* karr  = p; p += M * INNER;
  _Float16* vtr   = p; p += M * INNER;
  _Float16* attn  = qlin;  // q_lin dead after rope -> reuse

  dim3 tb(32, 8);
  transpose_to_f16<<<dim3(DIM / 32, INNER / 8), tb, 0, stream>>>(Wq, wqt, DIM, INNER);
  transpose_to_f16<<<dim3(DIM / 32, 2 * INNER / 8), tb, 0, stream>>>(Wkv, wkvt, DIM, 2 * INNER);
  transpose_to_f16<<<dim3(INNER / 32, DIM / 8), tb, 0, stream>>>(Wo, wot, INNER, DIM);

  rmsnorm_f16<<<(int)M, 256, 0, stream>>>(x, gamma, xn);

  gemm_lds<true><<<dim3((int)(M / 128), INNER / 64), 256, 0, stream>>>(
      xn, wqt, qlin, nullptr, nullptr, (int)M, INNER, DIM);
  gemm_lds<true><<<dim3((int)(M / 128), 2 * INNER / 64), 256, 0, stream>>>(
      xn, wkvt, kvlin, nullptr, nullptr, (int)M, 2 * INNER, DIM);

  const int elems = BATCH * HEADS * NTOK * DHEAD;  // 4M
  rope_head<<<elems / 256, 256, 0, stream>>>(qlin, freqs, qarr, 0, INNER, 0.125f);
  rope_head<<<elems / 256, 256, 0, stream>>>(kvlin, freqs, karr, 0, 2 * INNER, 1.0f);
  v_transpose<<<elems / 256, 256, 0, stream>>>(kvlin, vtr);

  attn_fa<<<dim3(NTOK / 16, HEADS, BATCH), 32, 0, stream>>>(qarr, karr, vtr, attn);

  gemm_lds<false><<<dim3((int)(M / 128), DIM / 64), 256, 0, stream>>>(
      attn, wot, nullptr, out, bo, (int)M, DIM, INNER);
}
